// GatedConvolution_1065151889995
// MI455X (gfx1250) — matrix operand order repaired and checked by reading, not compiled, not run on hardware
//
#include <hip/hip_runtime.h>

typedef __bf16 bf16;
typedef __bf16 v16bf __attribute__((ext_vector_type(16)));
typedef __bf16 v8bf  __attribute__((ext_vector_type(8)));
typedef float  v8f   __attribute__((ext_vector_type(8)));

// Problem constants
constexpr int CB = 16;        // batch
constexpr int CN = 512;       // nodes
constexpr int CH = 512;       // hid dim
constexpr int CD = 4;         // directions
constexpr int CM = CB * CN;   // 8192 flat rows
constexpr int CDH = CD * CH;  // 2048
constexpr int CG = 3 * CH;    // 1536 (r|z|n concat)

// ---------------- workspace layout (bytes) ----------------
constexpr size_t OFF_HIDB  = 0;                                   // bf16 [B*N, H]
constexpr size_t OFF_HIDBT = OFF_HIDB  + (size_t)CM * CH * 2;     // bf16 [B][H][N]
constexpr size_t OFF_WDB   = OFF_HIDBT + (size_t)CM * CH * 2;     // bf16 [H, D*H]   (= Wd layout)
constexpr size_t OFF_WIB   = OFF_WDB   + (size_t)CH * CDH * 2;    // bf16 [3H, H]    (Wri;Wzi;Wni)
constexpr size_t OFF_WHB   = OFF_WIB   + (size_t)CG * CH * 2;     // bf16 [3H, H]    (Wrh;Wzh;Wnh)
constexpr size_t OFF_BIC   = OFF_WHB   + (size_t)CG * CH * 2;     // f32  [3H]
constexpr size_t OFF_BHC   = OFF_BIC   + (size_t)CG * 4;          // f32  [3H]
constexpr size_t OFF_WGT   = OFF_BHC   + (size_t)CG * 4;          // bf16 [B][D*N, N]
constexpr size_t OFF_AGGB  = OFF_WGT   + (size_t)CB * CDH * CN * 2; // bf16 [B*N, D*H]
constexpr size_t OFF_OUTB  = OFF_AGGB  + (size_t)CM * CDH * 2;    // bf16 [B*N, H]
constexpr size_t OFF_G1    = OFF_OUTB  + (size_t)CM * CH * 2;     // f32  [B*N, 3H]
constexpr size_t OFF_G2    = OFF_G1    + (size_t)CM * CG * 4;     // f32  [B*N, 3H]
// total ~189 MB

// ---------------- prep kernels ----------------

// f32 -> bf16 flat convert
__global__ __launch_bounds__(256) void cvt_bf16(const float* __restrict__ src,
                                                bf16* __restrict__ dst, int n) {
  int idx = blockIdx.x * 256 + threadIdx.x;
  if (idx < n) dst[idx] = (bf16)src[idx];
}

// hid -> bf16 (row-major) + bf16 transposed per batch: hbT[b][h][m]
__global__ __launch_bounds__(256) void cvt_hid(const float* __restrict__ hid,
                                               bf16* __restrict__ hb,
                                               bf16* __restrict__ hbT) {
  size_t idx = (size_t)blockIdx.x * 256 + threadIdx.x;   // over B*N*H = 4194304
  float v = hid[idx];
  bf16 bv = (bf16)v;
  hb[idx] = bv;
  int b   = (int)(idx >> 18);          // N*H = 2^18
  int rem = (int)(idx & 262143);
  int m = rem >> 9;                    // H = 2^9
  int h = rem & 511;
  hbT[((size_t)b << 18) + ((size_t)h << 9) + m] = bv;
}

// concat three 512-long f32 biases
__global__ void cat3f(const float* __restrict__ a, const float* __restrict__ b,
                      const float* __restrict__ c, float* __restrict__ dst) {
  int idx = blockIdx.x * 256 + threadIdx.x;
  if (idx < CH) {
    dst[idx]          = a[idx];
    dst[CH + idx]     = b[idx];
    dst[2 * CH + idx] = c[idx];
  }
}

// Build dense bf16 aggregation weight matrix:
//   Wgt[b][(d*N + n)][m] = (adj[b,n,m]==d+1) ? 1/(cnt_d + 1e-8) : 0
__global__ __launch_bounds__(256) void build_wgt(const int* __restrict__ adj,
                                                 bf16* __restrict__ Wgt) {
  const int n = blockIdx.x;
  const int b = blockIdx.y;
  const int* row = adj + ((size_t)b * CN + n) * CN;
  __shared__ int cnt[CD];
  if (threadIdx.x < CD) cnt[threadIdx.x] = 0;
  __syncthreads();
  int a0 = row[threadIdx.x];
  int a1 = row[threadIdx.x + 256];
  if (a0 >= 1 && a0 <= CD) atomicAdd(&cnt[a0 - 1], 1);
  if (a1 >= 1 && a1 <= CD) atomicAdd(&cnt[a1 - 1], 1);
  __syncthreads();
  float inv[CD];
#pragma unroll
  for (int d = 0; d < CD; ++d) inv[d] = 1.0f / ((float)cnt[d] + 1e-8f);
  bf16* base = Wgt + (size_t)b * CDH * CN;
#pragma unroll
  for (int d = 0; d < CD; ++d) {
    size_t roff = (size_t)(d * CN + n) * CN;
    base[roff + threadIdx.x]       = (a0 == d + 1) ? (bf16)inv[d] : (bf16)0.0f;
    base[roff + threadIdx.x + 256] = (a1 == d + 1) ? (bf16)inv[d] : (bf16)0.0f;
  }
}

// ---------------- WMMA GEMM (async-LDS double-buffered) ----------------
// C[M,Ncols] = A[M,K] (bf16 row-major, lda) * B[K,Ncols] with B[k,j] = BT[j,k]
// (BT bf16 row-major [Ncols,K], ldb). f32 accumulate via v_wmma_f32_16x16x32_bf16.
// Block: 256 threads = 8 waves as 4(M)x2(N); wave tile 32x32; block tile 128x64.
// Tiles staged to LDS with global_load_async_to_lds_b128 (ASYNCcnt), double buffered.
// MODE 0: store bf16 with agg row remap (r=d*N+n -> [n, d*H+col]), per-batch.
// MODE 1: bias + relu -> bf16.
// MODE 2: bias -> f32.

// LDS byte-offsets within the (single) shared allocation:
//   A stage s : s*8192      (128 rows x 32 bf16 = 64B/row)
//   B stage s : 16384 + s*4096  (64 rows x 32 bf16)
__device__ __forceinline__ void async_b128(unsigned lds_off, const bf16* g) {
  asm volatile("global_load_async_to_lds_b128 %0, %1, off"
               :: "v"(lds_off), "v"(g) : "memory");
}

template <int MODE>
__global__ __launch_bounds__(256) void wmma_gemm(
    const bf16* __restrict__ A, long long sA, int lda,
    const bf16* __restrict__ BT, long long sB, int ldb,
    const float* __restrict__ bias,
    void* __restrict__ C, long long sC, int ldc, int K) {
  __shared__ __align__(16) unsigned char smem[2 * 8192 + 2 * 4096];  // 24 KB

  const int t       = threadIdx.x;
  const int lane    = t & 31;
  const int wave    = t >> 5;
  const int laneRow = lane & 15;
  const int hi      = lane >> 4;            // 0: lanes 0-15, 1: lanes 16-31
  const int blockM  = blockIdx.y * 128;
  const int blockN  = blockIdx.x * 64;
  const int mLoc    = (wave >> 1) * 32;     // wave tile origin inside block tile
  const int nLoc    = (wave & 1) * 32;
  A  += (size_t)blockIdx.z * sA;
  BT += (size_t)blockIdx.z * sB;

  v8f acc[2][2];
#pragma unroll
  for (int mi = 0; mi < 2; ++mi)
#pragma unroll
    for (int ni = 0; ni < 2; ++ni)
#pragma unroll
      for (int i = 0; i < 8; ++i) acc[mi][ni][i] = 0.0f;

  // cooperative async prefetch of one K-step tile pair into stage s
  auto prefetch = [&](int k0, int s) {
    // A tile: 512 x 16B chunks; 2 per thread. chunk c -> row c>>2, col-bytes (c&3)*16
#pragma unroll
    for (int j = 0; j < 2; ++j) {
      int c = t + j * 256;
      const bf16* g = A + (size_t)(blockM + (c >> 2)) * lda + k0 + (c & 3) * 8;
      async_b128((unsigned)(s * 8192 + c * 16), g);
    }
    // B tile: 256 x 16B chunks; 1 per thread
    {
      const bf16* g = BT + (size_t)(blockN + (t >> 2)) * ldb + k0 + (t & 3) * 8;
      async_b128((unsigned)(16384 + s * 4096 + t * 16), g);
    }
  };

  const int nk = K >> 5;
  prefetch(0, 0);
  for (int i = 0; i < nk; ++i) {
    const int s = i & 1;
    if (i + 1 < nk) {
      prefetch((i + 1) << 5, s ^ 1);
      asm volatile("s_wait_asynccnt 0x3" ::: "memory");  // own stage-i ops done
    } else {
      asm volatile("s_wait_asynccnt 0x0" ::: "memory");
    }
    __syncthreads();  // all waves' stage-i tile visible

    const bf16* As = (const bf16*)(smem + s * 8192);
    const bf16* Bs = (const bf16*)(smem + 16384 + s * 4096);

    v16bf af[2], bf_[2];
    // A frag (16x32): lanes 0-15 hold K {0..7,16..23}, lanes 16-31 {8..15,24..31}
#pragma unroll
    for (int mi = 0; mi < 2; ++mi) {
      const bf16* ap = As + (size_t)(mLoc + mi * 16 + laneRow) * 32;
      v8bf x = *(const v8bf*)(ap + hi * 8);
      v8bf y = *(const v8bf*)(ap + 16 + hi * 8);
      af[mi] = __builtin_shufflevector(x, y, 0,1,2,3,4,5,6,7,8,9,10,11,12,13,14,15);
    }
    // B frag (32x16): lanes 0-15 hold K {0..7,16..23}, lanes 16-31 {8..15,24..31}; N = laneRow
#pragma unroll
    for (int ni = 0; ni < 2; ++ni) {
      const bf16* bp = Bs + (size_t)(nLoc + ni * 16 + laneRow) * 32;
      v8bf x = *(const v8bf*)(bp + hi * 8);
      v8bf y = *(const v8bf*)(bp + 16 + hi * 8);
      bf_[ni] = __builtin_shufflevector(x, y, 0,1,2,3,4,5,6,7,8,9,10,11,12,13,14,15);
    }
#pragma unroll
    for (int mi = 0; mi < 2; ++mi)
#pragma unroll
      for (int ni = 0; ni < 2; ++ni)
        acc[mi][ni] = __builtin_amdgcn_wmma_f32_16x16x32_bf16(
            false, af[mi], false, bf_[ni], (short)0, acc[mi][ni], false, false);

    __syncthreads();  // done reading stage s before iter i+1 overwrites it
  }

  // Epilogue. C/D layout: elem i -> row blockM+mLoc+mi*16+hi*8+i, col blockN+nLoc+ni*16+laneRow
#pragma unroll
  for (int mi = 0; mi < 2; ++mi)
#pragma unroll
    for (int ni = 0; ni < 2; ++ni) {
      const int col = blockN + nLoc + ni * 16 + laneRow;
#pragma unroll
      for (int i = 0; i < 8; ++i) {
        const int row = blockM + mLoc + mi * 16 + hi * 8 + i;
        float v = acc[mi][ni][i];
        if (MODE == 0) {
          bf16* cp = (bf16*)C + (size_t)blockIdx.z * sC;
          int d = row >> 9, n = row & 511;               // row = d*N + n
          cp[(size_t)n * CDH + d * CH + col] = (bf16)v;
        } else if (MODE == 1) {
          float tv = v + bias[col];
          tv = tv > 0.0f ? tv : 0.0f;
          ((bf16*)C)[(size_t)row * ldc + col] = (bf16)tv;
        } else {
          ((float*)C)[(size_t)row * ldc + col] = v + bias[col];
        }
      }
    }
}

// ---------------- fused gates + LayerNorm ----------------
__global__ __launch_bounds__(256) void gate_ln(
    const float* __restrict__ G1, const float* __restrict__ G2,
    const float* __restrict__ hid, const float* __restrict__ gamma,
    const float* __restrict__ beta, float* __restrict__ out) {
  const int row = blockIdx.x;                 // 0..B*N-1
  const float* g1 = G1 + (size_t)row * CG;
  const float* g2 = G2 + (size_t)row * CG;
  const float* hv = hid + (size_t)row * CH;

  float o[2];
  float s = 0.0f, ss = 0.0f;
#pragma unroll
  for (int t = 0; t < 2; ++t) {
    int h = threadIdx.x + t * 256;
    float r  = 1.0f / (1.0f + expf(-(g1[h] + g2[h])));
    float z  = 1.0f / (1.0f + expf(-(g1[CH + h] + g2[CH + h])));
    float nn = tanhf(g1[2 * CH + h] + r * g2[2 * CH + h]);
    float ov = (1.0f - z) * nn + z * hv[h];
    o[t] = ov;
    s += ov;
    ss += ov * ov;
  }

  __shared__ float red[256];
  red[threadIdx.x] = s;
  __syncthreads();
  for (int off = 128; off > 0; off >>= 1) {
    if (threadIdx.x < off) red[threadIdx.x] += red[threadIdx.x + off];
    __syncthreads();
  }
  float total = red[0];
  __syncthreads();
  red[threadIdx.x] = ss;
  __syncthreads();
  for (int off = 128; off > 0; off >>= 1) {
    if (threadIdx.x < off) red[threadIdx.x] += red[threadIdx.x + off];
    __syncthreads();
  }
  float total2 = red[0];

  float mu   = total / (float)CH;
  float var  = total2 / (float)CH - mu * mu;
  float rstd = rsqrtf(var + 1e-5f);
#pragma unroll
  for (int t = 0; t < 2; ++t) {
    int h = threadIdx.x + t * 256;
    out[(size_t)row * CH + h] = (o[t] - mu) * rstd * gamma[h] + beta[h];
  }
}

// ---------------- launch ----------------
extern "C" void kernel_launch(void* const* d_in, const int* in_sizes, int n_in,
                              void* d_out, int out_size, void* d_ws, size_t ws_size,
                              hipStream_t stream) {
  (void)in_sizes; (void)n_in; (void)out_size; (void)ws_size;
  const int*   adj   = (const int*)  d_in[0];
  const float* hid   = (const float*)d_in[1];
  const float* Wd    = (const float*)d_in[2];
  const float* bd    = (const float*)d_in[3];
  const float* gamma = (const float*)d_in[4];
  const float* beta  = (const float*)d_in[5];
  const float* Wri = (const float*)d_in[6];  const float* bri = (const float*)d_in[7];
  const float* Wrh = (const float*)d_in[8];  const float* brh = (const float*)d_in[9];
  const float* Wzi = (const float*)d_in[10]; const float* bzi = (const float*)d_in[11];
  const float* Wzh = (const float*)d_in[12]; const float* bzh = (const float*)d_in[13];
  const float* Wni = (const float*)d_in[14]; const float* bni = (const float*)d_in[15];
  const float* Wnh = (const float*)d_in[16]; const float* bnh = (const float*)d_in[17];

  char* ws = (char*)d_ws;
  bf16*  hidb  = (bf16*) (ws + OFF_HIDB);
  bf16*  hidbT = (bf16*) (ws + OFF_HIDBT);
  bf16*  Wdb   = (bf16*) (ws + OFF_WDB);
  bf16*  Wib   = (bf16*) (ws + OFF_WIB);
  bf16*  Whb   = (bf16*) (ws + OFF_WHB);
  float* bic   = (float*)(ws + OFF_BIC);
  float* bhc   = (float*)(ws + OFF_BHC);
  bf16*  Wgt   = (bf16*) (ws + OFF_WGT);
  bf16*  aggb  = (bf16*) (ws + OFF_AGGB);
  bf16*  outb  = (bf16*) (ws + OFF_OUTB);
  float* G1    = (float*)(ws + OFF_G1);
  float* G2    = (float*)(ws + OFF_G2);

  // --- precision staging (f32 -> bf16) ---
  cvt_hid<<<CM * CH / 256, 256, 0, stream>>>(hid, hidb, hidbT);
  cvt_bf16<<<CH * CDH / 256, 256, 0, stream>>>(Wd, Wdb, CH * CDH);
  cvt_bf16<<<CH * CH / 256, 256, 0, stream>>>(Wri, Wib,               CH * CH);
  cvt_bf16<<<CH * CH / 256, 256, 0, stream>>>(Wzi, Wib + CH * CH,     CH * CH);
  cvt_bf16<<<CH * CH / 256, 256, 0, stream>>>(Wni, Wib + 2 * CH * CH, CH * CH);
  cvt_bf16<<<CH * CH / 256, 256, 0, stream>>>(Wrh, Whb,               CH * CH);
  cvt_bf16<<<CH * CH / 256, 256, 0, stream>>>(Wzh, Whb + CH * CH,     CH * CH);
  cvt_bf16<<<CH * CH / 256, 256, 0, stream>>>(Wnh, Whb + 2 * CH * CH, CH * CH);
  cat3f<<<2, 256, 0, stream>>>(bri, bzi, bni, bic);
  cat3f<<<2, 256, 0, stream>>>(brh, bzh, bnh, bhc);

  // --- dense directional-average matrix from adj ---
  build_wgt<<<dim3(CN, CB), 256, 0, stream>>>(adj, Wgt);

  // --- GEMM 1: agg[b] = Wgt[b] (2048x512) @ hid[b] (512x512), remap-store bf16 ---
  wmma_gemm<0><<<dim3(CH / 64, CDH / 128, CB), 256, 0, stream>>>(
      Wgt, (long long)CDH * CN, CN,
      hidbT, (long long)CH * CN, CN,
      nullptr, aggb, (long long)CN * CDH, 0, CN);

  // --- GEMM 2: output = relu(agg @ Wd.T + bd), [8192,2048]@[2048,512] -> bf16 ---
  wmma_gemm<1><<<dim3(CH / 64, CM / 128, 1), 256, 0, stream>>>(
      aggb, 0, CDH, Wdb, 0, CDH, bd, outb, 0, CH, CDH);

  // --- GEMM 3/4: gate pre-activations, [8192,512]@[512,1536] -> f32 ---
  wmma_gemm<2><<<dim3(CG / 64, CM / 128, 1), 256, 0, stream>>>(
      outb, 0, CH, Wib, 0, CH, bic, G1, 0, CG, CH);
  wmma_gemm<2><<<dim3(CG / 64, CM / 128, 1), 256, 0, stream>>>(
      hidb, 0, CH, Whb, 0, CH, bhc, G2, 0, CG, CH);

  // --- fused GRU gates + LayerNorm ---
  gate_ln<<<CM, 256, 0, stream>>>(G1, G2, hid, gamma, beta, (float*)d_out);
}
